// CBAMSA_76132590289611
// MI455X (gfx1250) — matrix operand order repaired and checked by reading, not compiled, not run on hardware
//
#include <hip/hip_runtime.h>

// ---------------- problem constants ----------------
#define Bb   4
#define Cc   256
#define Nn   4096      // H*W = 64*64
#define NHh  4
#define HDd  64
#define KDd  32
#define HQKV 512

typedef __attribute__((ext_vector_type(16))) _Float16 v16h;
typedef __attribute__((ext_vector_type(8)))  _Float16 v8h;
typedef __attribute__((ext_vector_type(8)))  float    v8f;

__device__ __forceinline__ v8h ld8h(const _Float16* p) { return *(const v8h*)p; }

__device__ __forceinline__ v16h cat16(v8h lo, v8h hi) {
  v16h r;
#pragma unroll
  for (int i = 0; i < 8; ++i) { r[i] = lo[i]; r[i + 8] = hi[i]; }
  return r;
}

__device__ __forceinline__ v8f wmma_f16(v16h a, v16h b, v8f c) {
  return __builtin_amdgcn_wmma_f32_16x16x32_f16(false, a, false, b, (short)0, c,
                                                false, false);
}

// ---------------- workspace layout (bytes) ----------------
constexpr size_t OFF_XHT   = 0;                                          // x transposed f16 [B][N][C]
constexpr size_t OFF_WQH   = OFF_XHT  + (size_t)Bb * Nn * Cc * 2;        // w_qkv f16 [512][256]
constexpr size_t OFF_WPH   = OFF_WQH  + (size_t)HQKV * Cc * 2;           // w_proj f16 [256][256]
constexpr size_t OFF_QT    = OFF_WPH  + (size_t)Cc * Cc * 2;             // q^T f16 [B][NH][N][KD] (pre-scaled)
constexpr size_t OFF_KT    = OFF_QT   + (size_t)Bb * NHh * Nn * KDd * 2; // k^T f16 [B][NH][N][KD]
constexpr size_t OFF_VH    = OFF_KT   + (size_t)Bb * NHh * Nn * KDd * 2; // v f16 [B][NH][HD][N]
constexpr size_t OFF_VF    = OFF_VH   + (size_t)Bb * NHh * HDd * Nn * 2; // v f32 [B][C][N]
constexpr size_t OFF_ATT   = OFF_VF   + (size_t)Bb * Cc * Nn * 4;        // attn out f32 [B][C][N]
constexpr size_t OFF_SUMT  = OFF_ATT  + (size_t)Bb * Cc * Nn * 4;        // (att+pe) f16 [B][N][C]
constexpr size_t OFF_XRES  = OFF_SUMT + (size_t)Bb * Nn * Cc * 2;        // residual f32 [B][C][N]
constexpr size_t OFF_AVG   = OFF_XRES + (size_t)Bb * Cc * Nn * 4;        // [B][C]
constexpr size_t OFF_MAXP  = OFF_AVG  + (size_t)Bb * Cc * 4;
constexpr size_t OFF_CA    = OFF_MAXP + (size_t)Bb * Cc * 4;
constexpr size_t OFF_SME   = OFF_CA   + (size_t)Bb * Cc * 4;             // [B][N]
constexpr size_t OFF_SMA   = OFF_SME  + (size_t)Bb * Nn * 4;

// ---------------- kernel 0a: weights f32 -> f16 ----------------
__global__ void k_convw(const float* wq, const float* wp, _Float16* wqh, _Float16* wph) {
  int i = blockIdx.x * 256 + threadIdx.x;
  if (i < HQKV * Cc) wqh[i] = (_Float16)wq[i];
  else {
    int j = i - HQKV * Cc;
    if (j < Cc * Cc) wph[j] = (_Float16)wp[j];
  }
}

// ---------------- kernel 0b: x[B][C][N] -> x_t f16 [B][N][C] ----------------
__global__ void k_transx(const float* x, _Float16* xt) {
  __shared__ float tile[32][33];
  int b = blockIdx.z, c0 = blockIdx.y * 32, n0 = blockIdx.x * 32;
  int tx = threadIdx.x, ty = threadIdx.y;
#pragma unroll
  for (int i = 0; i < 4; ++i) {
    int c = c0 + ty + i * 8;
    tile[ty + i * 8][tx] = x[((size_t)b * Cc + c) * Nn + n0 + tx];
  }
  __syncthreads();
#pragma unroll
  for (int i = 0; i < 4; ++i) {
    int n = n0 + ty + i * 8;
    xt[((size_t)b * Nn + n) * Cc + c0 + tx] = (_Float16)tile[tx][ty + i * 8];
  }
}

// ---------------- kernel 1: QKV GEMM (WMMA) + repack q^T/k^T/v ----------------
// Per wave: 16(o) x 16(n) output tile, K=256 in steps of 32.
__global__ void __launch_bounds__(128) k_qkv(const _Float16* wh, const _Float16* xt,
                                             const float* bqkv, _Float16* qt, _Float16* ktt,
                                             _Float16* vhh, float* vf) {
  const int tid = threadIdx.x, wv = tid >> 5, lane = tid & 31, row = lane & 15;
  const bool hi = lane >= 16;
  const int b = blockIdx.z;
  const int o0 = blockIdx.y * 16;
  const int n0 = (blockIdx.x * 4 + wv) * 16;
  const int n = n0 + row;

  const _Float16* arow = wh + (size_t)(o0 + row) * Cc;            // A: W[o][c]
  const _Float16* brow = xt + ((size_t)b * Nn + n) * Cc + (hi ? 8 : 0); // B: x_t[n][c]
  v8f acc = {0, 0, 0, 0, 0, 0, 0, 0};
#pragma unroll
  for (int c0 = 0; c0 < Cc; c0 += 32) {
    v16h A = cat16(ld8h(arow + c0 + (hi ? 8 : 0)),
                   ld8h(arow + c0 + 16 + (hi ? 8 : 0)));
    v16h Bm = cat16(ld8h(brow + c0), ld8h(brow + c0 + 16));
    acc = wmma_f16(A, Bm, acc);
  }
  float vals[8];
#pragma unroll
  for (int r = 0; r < 8; ++r) vals[r] = acc[r] + bqkv[o0 + r + (hi ? 8 : 0)];

  const int h = o0 >> 7;        // head = o / 128
  const int j0 = o0 & 127;      // offset within head block
  if (j0 < 64) {                // q (j<32) or k (32<=j<64): store transposed [n][kd]
    const bool isq = j0 < 32;
    const float scale = 0.17677669529663687f;  // KD^-0.5 folded into q
    const int jb = (isq ? j0 : j0 - 32) + (hi ? 8 : 0);
    _Float16* dst = (isq ? qt : ktt) + ((size_t)(b * NHh + h) * Nn + n) * KDd + jb;
    v8h pk;
#pragma unroll
    for (int r = 0; r < 8; ++r) pk[r] = (_Float16)(isq ? vals[r] * scale : vals[r]);
    *(v8h*)dst = pk;
  } else {                      // v: native [d][n] in f16 and f32
    const int d0 = j0 - 64 + (hi ? 8 : 0);
#pragma unroll
    for (int r = 0; r < 8; ++r) {
      const int d = d0 + r;
      vhh[((size_t)(b * NHh + h) * HDd + d) * Nn + n] = (_Float16)vals[r];
      vf[((size_t)b * Cc + h * 64 + d) * Nn + n] = vals[r];
    }
  }
}

// ---------------- kernel 2: flash attention (WMMA, S^T orientation) ----------------
// WG = 8 waves = 128 n-rows; each wave owns 16 n and 4 accum tiles (64 d).
// K/V tiles (32 m-columns) staged in LDS, shared by all waves.
__global__ void __launch_bounds__(256) k_flash(const _Float16* qt, const _Float16* kt,
                                               const _Float16* vh, float* att) {
  __shared__ __align__(16) _Float16 kts[32 * 32];  // [m_rel][k]
  __shared__ __align__(16) _Float16 vhs[64 * 32];  // [d][m_rel]
  const int tid = threadIdx.x, wv = tid >> 5, lane = tid & 31, row = lane & 15;
  const bool hi = lane >= 16;
  const int b = blockIdx.z, h = blockIdx.y;
  const int n0 = (blockIdx.x * 8 + wv) * 16;
  const int n = n0 + row;

  // Q fragment (B layout: lanes 0-15 hold k 0-7,16-23, lanes 16-31 hold k 8-15,24-31)
  const _Float16* qrow = qt + ((size_t)(b * NHh + h) * Nn + n) * KDd + (hi ? 8 : 0);
  const v16h Q = cat16(ld8h(qrow), ld8h(qrow + 16));

  const _Float16* ktg = kt + (size_t)(b * NHh + h) * Nn * KDd;
  const _Float16* vhg = vh + (size_t)(b * NHh + h) * HDd * Nn;

  v8f O0 = {0, 0, 0, 0, 0, 0, 0, 0}, O1 = O0, O2 = O0, O3 = O0;
  float run_max = -3.0e38f, run_sum = 0.0f;

  for (int m0 = 0; m0 < Nn; m0 += 32) {
    __syncthreads();
    // stage K^T tile: rows m0..m0+31 are contiguous (2 KB)
    ((uint2*)kts)[tid] = ((const uint2*)(ktg + (size_t)m0 * KDd))[tid];
    // stage V tile: 64 rows x 32 cols (4 KB)
    {
      const int d = tid >> 2, q4 = tid & 3;
      *(float4*)(vhs + d * 32 + q4 * 8) =
          *(const float4*)(vhg + (size_t)d * Nn + m0 + q4 * 8);
    }
    __syncthreads();
    if (m0 + 32 < Nn) __builtin_prefetch(ktg + (size_t)(m0 + 32) * KDd, 0, 1);

    // S^T = K^T x Q  (two 16m tiles)
    const _Float16* kr0 = kts + row * 32 + (hi ? 8 : 0);
    const _Float16* kr1 = kts + (row + 16) * 32 + (hi ? 8 : 0);
    const v16h A0 = cat16(ld8h(kr0), ld8h(kr0 + 16));
    const v16h A1 = cat16(ld8h(kr1), ld8h(kr1 + 16));
    v8f z = {0, 0, 0, 0, 0, 0, 0, 0};
    v8f s0 = wmma_f16(A0, Q, z);
    v8f s1 = wmma_f16(A1, Q, z);

    // online softmax over m (lane pair L / L^16 covers one n-column)
    float lmax = s0[0];
#pragma unroll
    for (int r = 1; r < 8; ++r) lmax = fmaxf(lmax, s0[r]);
#pragma unroll
    for (int r = 0; r < 8; ++r) lmax = fmaxf(lmax, s1[r]);
    lmax = fmaxf(lmax, __shfl_xor(lmax, 16, 32));
    const float nmax = fmaxf(run_max, lmax);
    const float corr = __expf(run_max - nmax);
    run_sum *= corr;
#pragma unroll
    for (int r = 0; r < 8; ++r) {
      O0[r] *= corr; O1[r] *= corr; O2[r] *= corr; O3[r] *= corr;
    }
    float p0[8], p1[8], psum = 0.0f;
#pragma unroll
    for (int r = 0; r < 8; ++r) {
      p0[r] = __expf(s0[r] - nmax);
      p1[r] = __expf(s1[r] - nmax);
      psum += p0[r] + p1[r];
    }
    run_sum += psum;
    run_max = nmax;

    // pack P into B-matrix layout (lane-local only; shfl results unused)
    union { unsigned u; _Float16 hx[2]; } pk;
    unsigned mp0[4], mp1[4];
#pragma unroll
    for (int j = 0; j < 4; ++j) {
      pk.hx[0] = (_Float16)p0[2 * j]; pk.hx[1] = (_Float16)p0[2 * j + 1]; mp0[j] = pk.u;
      pk.hx[0] = (_Float16)p1[2 * j]; pk.hx[1] = (_Float16)p1[2 * j + 1]; mp1[j] = pk.u;
    }
    union { v16h v; unsigned u[8]; } pb;
#pragma unroll
    for (int j = 0; j < 4; ++j) {
      const unsigned op0 = (unsigned)__shfl_xor((int)mp0[j], 16, 32);
      const unsigned op1 = (unsigned)__shfl_xor((int)mp1[j], 16, 32);
      pb.u[j]     = mp0[j];   // m 0..7  | 8..15
      pb.u[4 + j] = mp1[j];   // m 16..23 | 24..31
    }

    // O += V x P   (A = V tile 16d x 32m)
    const _Float16* v0 = vhs + (row) * 32 + (hi ? 8 : 0);
    const _Float16* v1 = vhs + (16 + row) * 32 + (hi ? 8 : 0);
    const _Float16* v2 = vhs + (32 + row) * 32 + (hi ? 8 : 0);
    const _Float16* v3 = vhs + (48 + row) * 32 + (hi ? 8 : 0);
    O0 = wmma_f16(cat16(ld8h(v0), ld8h(v0 + 16)), pb.v, O0);
    O1 = wmma_f16(cat16(ld8h(v1), ld8h(v1 + 16)), pb.v, O1);
    O2 = wmma_f16(cat16(ld8h(v2), ld8h(v2 + 16)), pb.v, O2);
    O3 = wmma_f16(cat16(ld8h(v3), ld8h(v3 + 16)), pb.v, O3);
  }

  const float tot = run_sum + __shfl_xor(run_sum, 16, 32);
  const float inv = 1.0f / tot;
  float* ab = att + ((size_t)b * Cc + h * 64) * Nn;
#pragma unroll
  for (int r = 0; r < 8; ++r) {
    const int dof = r + (hi ? 8 : 0);
    ab[(size_t)(dof) * Nn + n]      = O0[r] * inv;
    ab[(size_t)(16 + dof) * Nn + n] = O1[r] * inv;
    ab[(size_t)(32 + dof) * Nn + n] = O2[r] * inv;
    ab[(size_t)(48 + dof) * Nn + n] = O3[r] * inv;
  }
}

// ---------------- kernel 3: att + depthwise 3x3(v) + b_pe -> f16 [B][N][C] ----------------
__global__ void k_pesum(const float* att, const float* vf, const float* wpe,
                        const float* bpe, _Float16* st) {
  const int b = blockIdx.z, c = blockIdx.y;
  const int n = blockIdx.x * 256 + threadIdx.x;
  const int y = n >> 6, xw = n & 63;
  const float* vb = vf + ((size_t)b * Cc + c) * Nn;
  float acc = att[((size_t)b * Cc + c) * Nn + n] + bpe[c];
#pragma unroll
  for (int dy = -1; dy <= 1; ++dy) {
    const int yy = y + dy;
    if (yy < 0 || yy >= 64) continue;
#pragma unroll
    for (int dx = -1; dx <= 1; ++dx) {
      const int xx = xw + dx;
      if (xx < 0 || xx >= 64) continue;
      acc += wpe[c * 9 + (dy + 1) * 3 + (dx + 1)] * vb[yy * 64 + xx];
    }
  }
  st[((size_t)b * Nn + n) * Cc + c] = (_Float16)acc;
}

// ---------------- kernel 4: proj GEMM (WMMA) + bias + residual ----------------
__global__ void __launch_bounds__(128) k_proj(const _Float16* wh, const _Float16* st,
                                              const float* bproj, const float* x,
                                              float* xres) {
  const int tid = threadIdx.x, wv = tid >> 5, lane = tid & 31, row = lane & 15;
  const bool hi = lane >= 16;
  const int b = blockIdx.z;
  const int o0 = blockIdx.y * 16;
  const int n0 = (blockIdx.x * 4 + wv) * 16;
  const int n = n0 + row;

  const _Float16* arow = wh + (size_t)(o0 + row) * Cc;
  const _Float16* brow = st + ((size_t)b * Nn + n) * Cc + (hi ? 8 : 0);
  v8f acc = {0, 0, 0, 0, 0, 0, 0, 0};
#pragma unroll
  for (int c0 = 0; c0 < Cc; c0 += 32) {
    v16h A = cat16(ld8h(arow + c0 + (hi ? 8 : 0)),
                   ld8h(arow + c0 + 16 + (hi ? 8 : 0)));
    v16h Bm = cat16(ld8h(brow + c0), ld8h(brow + c0 + 16));
    acc = wmma_f16(A, Bm, acc);
  }
#pragma unroll
  for (int r = 0; r < 8; ++r) {
    const int o = o0 + r + (hi ? 8 : 0);
    const size_t idx = ((size_t)b * Cc + o) * Nn + n;
    xres[idx] = x[idx] + acc[r] + bproj[o];
  }
}

// ---------------- kernel 5: global avg/max pool over N per (b,c) ----------------
__global__ void k_pool(const float* xres, float* avg, float* mxp) {
  const int c = blockIdx.x, b = blockIdx.y, tid = threadIdx.x;
  const float* p = xres + ((size_t)b * Cc + c) * Nn;
  float s = 0.0f, m = -3.0e38f;
  for (int i = tid; i < Nn; i += 256) { const float v = p[i]; s += v; m = fmaxf(m, v); }
  __shared__ float ss[256], sm[256];
  ss[tid] = s; sm[tid] = m;
  __syncthreads();
  for (int stp = 128; stp > 0; stp >>= 1) {
    if (tid < stp) { ss[tid] += ss[tid + stp]; sm[tid] = fmaxf(sm[tid], sm[tid + stp]); }
    __syncthreads();
  }
  if (tid == 0) { avg[b * Cc + c] = ss[0] * (1.0f / Nn); mxp[b * Cc + c] = sm[0]; }
}

// ---------------- kernel 6: channel attention MLP ----------------
__global__ void k_ca(const float* avg, const float* mxp, const float* wfc1,
                     const float* wfc2, float* ca) {
  const int b = blockIdx.x, tid = threadIdx.x;
  __shared__ float sav[256], smx[256], ha[16], hm[16];
  sav[tid] = avg[b * Cc + tid];
  smx[tid] = mxp[b * Cc + tid];
  __syncthreads();
  if (tid < 16) {
    float s1 = 0.0f, s2 = 0.0f;
    for (int c = 0; c < Cc; ++c) {
      const float w = wfc1[tid * Cc + c];
      s1 += w * sav[c]; s2 += w * smx[c];
    }
    ha[tid] = fmaxf(s1, 0.0f);
    hm[tid] = fmaxf(s2, 0.0f);
  }
  __syncthreads();
  float s = 0.0f;
#pragma unroll
  for (int j = 0; j < 16; ++j) s += wfc2[tid * 16 + j] * (ha[j] + hm[j]);
  ca[b * Cc + tid] = 1.0f / (1.0f + __expf(-s));
}

// ---------------- kernel 7: per-pixel mean/max over channels of xres*ca ----------------
__global__ void k_samap(const float* xres, const float* ca, float* sme, float* sma) {
  const int b = blockIdx.y;
  const int n = blockIdx.x * 256 + threadIdx.x;
  float s = 0.0f, m = -3.0e38f;
  for (int c = 0; c < Cc; ++c) {
    const float v = xres[((size_t)b * Cc + c) * Nn + n] * ca[b * Cc + c];
    s += v; m = fmaxf(m, v);
  }
  sme[b * Nn + n] = s * (1.0f / Cc);
  sma[b * Nn + n] = m;
}

// ---------------- kernel 8: spatial-attention 3x3 conv + final scale ----------------
__global__ void k_final(const float* xres, const float* ca, const float* sme,
                        const float* sma, const float* wsa, float* out) {
  const int b = blockIdx.y;
  const int n = blockIdx.x * 256 + threadIdx.x;
  const int y = n >> 6, xw = n & 63;
  float acc = 0.0f;
#pragma unroll
  for (int dy = -1; dy <= 1; ++dy) {
    const int yy = y + dy;
    if (yy < 0 || yy >= 64) continue;
#pragma unroll
    for (int dx = -1; dx <= 1; ++dx) {
      const int xx = xw + dx;
      if (xx < 0 || xx >= 64) continue;
      const int k = (dy + 1) * 3 + (dx + 1);
      acc += wsa[k] * sme[b * Nn + yy * 64 + xx] + wsa[9 + k] * sma[b * Nn + yy * 64 + xx];
    }
  }
  const float sa = 1.0f / (1.0f + __expf(-acc));
  for (int c = 0; c < Cc; ++c) {
    const size_t idx = ((size_t)b * Cc + c) * Nn + n;
    out[idx] = xres[idx] * ca[b * Cc + c] * sa;
  }
}

// ---------------- launch ----------------
extern "C" void kernel_launch(void* const* d_in, const int* in_sizes, int n_in,
                              void* d_out, int out_size, void* d_ws, size_t ws_size,
                              hipStream_t stream) {
  const float* x     = (const float*)d_in[0];
  const float* wqkv  = (const float*)d_in[1];
  const float* bqkv  = (const float*)d_in[2];
  const float* wproj = (const float*)d_in[3];
  const float* bproj = (const float*)d_in[4];
  const float* wpe   = (const float*)d_in[5];
  const float* bpe   = (const float*)d_in[6];
  const float* wfc1  = (const float*)d_in[7];
  const float* wfc2  = (const float*)d_in[8];
  const float* wsa   = (const float*)d_in[9];

  char* ws = (char*)d_ws;
  _Float16* xt   = (_Float16*)(ws + OFF_XHT);
  _Float16* wqh  = (_Float16*)(ws + OFF_WQH);
  _Float16* wph  = (_Float16*)(ws + OFF_WPH);
  _Float16* qt   = (_Float16*)(ws + OFF_QT);
  _Float16* ktt  = (_Float16*)(ws + OFF_KT);
  _Float16* vhh  = (_Float16*)(ws + OFF_VH);
  float*    vf   = (float*)(ws + OFF_VF);
  float*    att  = (float*)(ws + OFF_ATT);
  _Float16* st   = (_Float16*)(ws + OFF_SUMT);
  float*    xres = (float*)(ws + OFF_XRES);
  float*    avg  = (float*)(ws + OFF_AVG);
  float*    mxp  = (float*)(ws + OFF_MAXP);
  float*    cap  = (float*)(ws + OFF_CA);
  float*    sme  = (float*)(ws + OFF_SME);
  float*    sma  = (float*)(ws + OFF_SMA);

  k_convw <<<768, 256, 0, stream>>>(wqkv, wproj, wqh, wph);
  k_transx<<<dim3(Nn / 32, Cc / 32, Bb), dim3(32, 8), 0, stream>>>(x, xt);
  k_qkv   <<<dim3(Nn / 64, HQKV / 16, Bb), 128, 0, stream>>>(wqh, xt, bqkv, qt, ktt, vhh, vf);
  k_flash <<<dim3(Nn / 128, NHh, Bb), 256, 0, stream>>>(qt, ktt, vhh, att);
  k_pesum <<<dim3(Nn / 256, Cc, Bb), 256, 0, stream>>>(att, vf, wpe, bpe, st);
  k_proj  <<<dim3(Nn / 64, Cc / 16, Bb), 128, 0, stream>>>(wph, st, bproj, x, xres);
  k_pool  <<<dim3(Cc, Bb), 256, 0, stream>>>(xres, avg, mxp);
  k_ca    <<<Bb, 256, 0, stream>>>(avg, mxp, wfc1, wfc2, cap);
  k_samap <<<dim3(Nn / 256, Bb), 256, 0, stream>>>(xres, cap, sme, sma);
  k_final <<<dim3(Nn / 256, Bb), 256, 0, stream>>>(xres, cap, sme, sma, wsa, (float*)d_out);
}
